// MCR2_HGPD_62680752718518
// MI455X (gfx1250) — compile-verified, matrix operand order checked by reading, not run on hardware
//
#include <hip/hip_runtime.h>
#include <hip/hip_bf16.h>

// ---------------------------------------------------------------------------
// GNN forward for gfx1250 (MI455X).
//  * GEMMs via v_wmma_f32_16x16x32_bf16, A fragments held in VGPRs across both
//    heads, packed B staged into LDS once per block through the CDNA5 async
//    Global->LDS DMA path (global_load_async_to_lds_b128 / s_wait_asynccnt).
//  * Edge aggregation via f32 global atomics; xw intermediate kept in bf16 to
//    halve the dominant gather traffic; global_prefetch_b8 lookahead on the
//    random xw-row gathers.
// ---------------------------------------------------------------------------

typedef __attribute__((ext_vector_type(8)))  __bf16 v8bf;
typedef __attribute__((ext_vector_type(16))) __bf16 v16bf;
typedef __attribute__((ext_vector_type(8)))  float  v8f;

#define U_TOT  100000
#define U_1    50000
#define E_TOT  1000000
#define H_DIM  128
#define DC_DIM 64
#define ED_DIM 85
#define ED_PAD 96
#define NHEAD  2
#define MT_TOT 6250   // U_TOT / 16
#define NT_TOT 8      // H_DIM / 16
#define KT_ED  3      // ED_PAD / 32
#define KT_DC  2      // DC_DIM / 32

static constexpr size_t alignup(size_t x) { return (x + 255) & ~size_t(255); }
static constexpr size_t OFF_DEG  = 0;                                          // U f32 (becomes dis)
static constexpr size_t OFF_CNT  = alignup(OFF_DEG + size_t(U_TOT) * 4);       // U f32
static constexpr size_t OFF_AGG  = alignup(OFF_CNT + size_t(U_TOT) * 4);       // U*64 f32
static constexpr size_t OFF_BNS  = alignup(OFF_AGG + size_t(U_TOT) * DC_DIM * 4); // 128 f32
static constexpr size_t OFF_BNS2 = OFF_BNS + 512;                              // 128 f32
static constexpr size_t ZERO_END = OFF_BNS2 + 512;
static constexpr size_t OFF_UF16 = alignup(ZERO_END);                          // U*96 bf16
static constexpr size_t OFF_AG16 = alignup(OFF_UF16 + size_t(U_TOT) * ED_PAD * 2); // U*64 bf16
static constexpr size_t OFF_XW16 = alignup(OFF_AG16 + size_t(U_TOT) * DC_DIM * 2); // 2*U*128 bf16
static constexpr size_t OFF_PRE  = alignup(OFF_XW16 + size_t(NHEAD) * U_TOT * H_DIM * 2); // 2*U*128 f32
static constexpr size_t OFF_BG   = alignup(OFF_PRE  + size_t(NHEAD) * U_TOT * H_DIM * 4);
static constexpr size_t OFF_BR   = alignup(OFF_BG + size_t(NHEAD) * NT_TOT * KT_ED * 512 * 2);
static constexpr size_t OFF_BL   = alignup(OFF_BR + size_t(NHEAD) * NT_TOT * KT_ED * 512 * 2);

__device__ __forceinline__ unsigned short f2bf(float f) {
    unsigned int u = __float_as_uint(f);
    unsigned int r = (u + 0x7FFFu + ((u >> 16) & 1u)) >> 16;  // RNE
    return (unsigned short)r;
}
__device__ __forceinline__ float bf2f(unsigned short u) {
    return __uint_as_float(((unsigned int)u) << 16);
}

// A fragment (16x32 bf16): lane half selects K offset {0,8}; elems 0..7 -> K+0..7,
// elems 8..15 -> K+16..23 (per ISA 16-bit A layout). Two 16B loads + shuffle.
__device__ __forceinline__ v16bf load_afrag(const unsigned short* p) {
    v8bf lo = *(const v8bf*)p;
    v8bf hi = *(const v8bf*)(p + 16);
    return __builtin_shufflevector(lo, hi, 0, 1, 2, 3, 4, 5, 6, 7,
                                   8, 9, 10, 11, 12, 13, 14, 15);
}

__device__ __forceinline__ v8f wmma_bf16(v16bf a, v16bf b, v8f c) {
    return __builtin_amdgcn_wmma_f32_16x16x32_bf16(false, a, false, b,
                                                   (short)0, c, false, false);
}

// CDNA5 async Global -> LDS copy of one 16B chunk (per-lane), tracked by ASYNCcnt.
__device__ __forceinline__ void async_g2lds_b128(unsigned lds_off, const void* gptr) {
    asm volatile("global_load_async_to_lds_b128 %0, %1, off"
                 :: "v"(lds_off), "v"(gptr) : "memory");
}
__device__ __forceinline__ void wait_asynccnt0() {
    asm volatile("s_wait_asynccnt 0" ::: "memory");
}

// --------------------------- kernels ---------------------------------------

__global__ void k_zero(float* __restrict__ p, int n) {
    int t = blockIdx.x * blockDim.x + threadIdx.x;
    if (t < n) p[t] = 0.f;
}

__global__ void k_build_userF(const float* __restrict__ u_feature,
                              const int* __restrict__ no_Nidx,
                              const float* __restrict__ emb_table,
                              const float* __restrict__ e0, const float* __restrict__ e3,
                              const float* __restrict__ e7, const float* __restrict__ e8,
                              const float* __restrict__ e9,
                              unsigned short* __restrict__ uf16) {
    int i = blockIdx.x * blockDim.x + threadIdx.x;
    if (i >= U_TOT) return;
    unsigned short* dst = uf16 + (size_t)i * ED_PAD;
    if (i < U_1) {
        const float* uf = u_feature + (size_t)i * 10;
        int i0 = (int)uf[0], i3 = (int)uf[3], i7 = (int)uf[7];
        int i8 = (int)uf[8], i9 = (int)uf[9];
        #pragma unroll
        for (int k = 0; k < 16; ++k) dst[k] = f2bf(e0[i0 * 16 + k]);
        dst[16] = f2bf(uf[1]);
        dst[17] = f2bf(uf[2]);
        #pragma unroll
        for (int k = 0; k < 16; ++k) dst[18 + k] = f2bf(e3[i3 * 16 + k]);
        dst[34] = f2bf(uf[4]);
        dst[35] = f2bf(uf[5]);
        dst[36] = f2bf(uf[6]);
        #pragma unroll
        for (int k = 0; k < 16; ++k) dst[37 + k] = f2bf(e7[i7 * 16 + k]);
        #pragma unroll
        for (int k = 0; k < 16; ++k) dst[53 + k] = f2bf(e8[i8 * 16 + k]);
        #pragma unroll
        for (int k = 0; k < 16; ++k) dst[69 + k] = f2bf(e9[i9 * 16 + k]);
    } else {
        int s = no_Nidx[i - U_1];
        const float* src = emb_table + (size_t)s * ED_DIM;
        for (int k = 0; k < ED_DIM; ++k) dst[k] = f2bf(src[k]);
    }
    #pragma unroll
    for (int k = ED_DIM; k < ED_PAD; ++k) dst[k] = 0;  // K padding
}

// Pack weights [h][K][128] into fragment-major bf16: dst[h][nt][kt][lane][e]
// matching B 32x16 layout: k = kt*32 + (lane>=16? 8:0) + (e<8? e : e+8), n = nt*16 + lane%16.
__global__ void k_packB(const float* __restrict__ w, unsigned short* __restrict__ dst,
                        int K, int KT, int total) {
    int idx = blockIdx.x * blockDim.x + threadIdx.x;
    if (idx >= total) return;
    int e    = idx & 15;
    int lane = (idx >> 4) & 31;
    int kt   = (idx >> 9) % KT;
    int nt   = ((idx >> 9) / KT) % NT_TOT;
    int h    = (idx >> 9) / (KT * NT_TOT);
    int k = kt * 32 + ((lane >> 4) << 3) + (e < 8 ? e : e + 8);
    int n = (nt << 4) + (lane & 15);
    float v = (k < K) ? w[((size_t)h * K + k) * H_DIM + n] : 0.f;
    dst[idx] = f2bf(v);
}

__global__ void k_degcnt(const int* __restrict__ uu_col, const int* __restrict__ cu_dst,
                         float* __restrict__ deg, float* __restrict__ cnt) {
    int e = blockIdx.x * blockDim.x + threadIdx.x;
    if (e >= E_TOT) return;
    atomicAdd(deg + uu_col[e], 1.f);
    atomicAdd(cnt + cu_dst[e], 1.f);
}

__global__ void k_dis(float* __restrict__ deg) {
    int i = blockIdx.x * blockDim.x + threadIdx.x;
    if (i >= U_TOT) return;
    float d = deg[i];
    deg[i] = (d > 0.f) ? rsqrtf(fmaxf(d, 1.f)) : 0.f;  // in-place: deg -> dis
}

__global__ void k_agg_scatter(const int* __restrict__ cu_src, const int* __restrict__ cu_dst,
                              const float* __restrict__ comment_x, float* __restrict__ agg) {
    int t = blockIdx.x * blockDim.x + threadIdx.x;
    if (t >= E_TOT * 16) return;
    int e = t >> 4;
    int j = (t & 15) << 2;
    int s = cu_src[e];
    int d = cu_dst[e];
    if (e + 16 < E_TOT)  // global_prefetch_b8 on the upcoming gather line
        __builtin_prefetch(comment_x + (size_t)cu_src[e + 16] * DC_DIM + j, 0, 1);
    float4 v = *(const float4*)(comment_x + (size_t)s * DC_DIM + j);
    float* dp = agg + (size_t)d * DC_DIM + j;
    atomicAdd(dp + 0, v.x);
    atomicAdd(dp + 1, v.y);
    atomicAdd(dp + 2, v.z);
    atomicAdd(dp + 3, v.w);
}

__global__ void k_agg_fin(const float* __restrict__ agg, const float* __restrict__ cnt,
                          unsigned short* __restrict__ ag16) {
    int t = blockIdx.x * blockDim.x + threadIdx.x;
    if (t >= U_TOT * DC_DIM) return;
    int i = t >> 6;
    float c = fmaxf(cnt[i], 1.f);
    ag16[t] = f2bf(agg[t] / c);
}

// Fused GEMM, both heads: per 16x16 tile and head h,
//   xw  = userF @ gcn_w[h]                         (3 WMMAs)
//   pre = userF @ sage_r_w[h] + agg @ sage_l_w[h]  (+ biases)  (3+2 WMMAs)
// A fragments loaded once per M-tile and reused for both heads; packed B for
// the current head staged into 64 KB LDS via async Global->LDS DMA.
__global__ void __launch_bounds__(256) k_gemm(
        const unsigned short* __restrict__ uf16,
        const unsigned short* __restrict__ ag16,
        const unsigned short* __restrict__ Bg,
        const unsigned short* __restrict__ Br,
        const unsigned short* __restrict__ Bl,
        const float* __restrict__ gcn_b,
        const float* __restrict__ sage_l_b,
        unsigned short* __restrict__ xw16,
        float* __restrict__ pre) {
    __shared__ unsigned short sB[32768];  // 64 KB: Bg[0..12287] | Br[12288..24575] | Bl[24576..32767]
    const int wave = threadIdx.x >> 5;
    const int lane = threadIdx.x & 31;
    const int mt = blockIdx.x * 8 + wave;
    const bool active = (mt < MT_TOT);          // wave-uniform
    const int m0 = (active ? mt : MT_TOT - 1) << 4;
    const int arow = m0 + (lane & 15);
    const int khalf = (lane >> 4) << 3;

    v16bf aU[KT_ED];
    {
        const unsigned short* rp = uf16 + (size_t)arow * ED_PAD + khalf;
        #pragma unroll
        for (int kt = 0; kt < KT_ED; ++kt) aU[kt] = load_afrag(rp + kt * 32);
    }
    v16bf aA[KT_DC];
    {
        const unsigned short* rp = ag16 + (size_t)arow * DC_DIM + khalf;
        #pragma unroll
        for (int kt = 0; kt < KT_DC; ++kt) aA[kt] = load_afrag(rp + kt * 32);
    }

    // LDS byte offset of sB (generic LDS pointers carry the LDS offset in addr[31:0])
    const unsigned sBase = (unsigned)(size_t)&sB[0];
    const int drow0 = m0 + ((lane >> 4) << 3);  // C/D layout: M = v + 8*(lane>=16)

    for (int h = 0; h < NHEAD; ++h) {
        {   // stage this head's packed B (4096 x 16B chunks) via async DMA
            const char* srcG = (const char*)(Bg + (size_t)h * (NT_TOT * KT_ED * 512));
            const char* srcR = (const char*)(Br + (size_t)h * (NT_TOT * KT_ED * 512));
            const char* srcL = (const char*)(Bl + (size_t)h * (NT_TOT * KT_DC * 512));
            #pragma unroll 1
            for (int c = threadIdx.x; c < 4096; c += 256) {
                const char* g;
                if (c < 1536)      g = srcG + (size_t)c * 16;
                else if (c < 3072) g = srcR + (size_t)(c - 1536) * 16;
                else               g = srcL + (size_t)(c - 3072) * 16;
                async_g2lds_b128(sBase + (unsigned)c * 16u, g);
            }
            wait_asynccnt0();
        }
        __syncthreads();
        if (active) {
            const unsigned short* sG = sB + lane * 16;
            const unsigned short* sR = sB + 12288 + lane * 16;
            const unsigned short* sL = sB + 24576 + lane * 16;
            const size_t hb = (size_t)h * U_TOT * H_DIM;
            #pragma unroll 1
            for (int nt = 0; nt < NT_TOT; ++nt) {
                v8f accx = {};
                v8f accp = {};
                #pragma unroll
                for (int kt = 0; kt < KT_ED; ++kt) {
                    v16bf bg = *(const v16bf*)(sG + (nt * KT_ED + kt) * 512);
                    accx = wmma_bf16(aU[kt], bg, accx);
                    v16bf br = *(const v16bf*)(sR + (nt * KT_ED + kt) * 512);
                    accp = wmma_bf16(aU[kt], br, accp);
                }
                #pragma unroll
                for (int kt = 0; kt < KT_DC; ++kt) {
                    v16bf bl = *(const v16bf*)(sL + (nt * KT_DC + kt) * 512);
                    accp = wmma_bf16(aA[kt], bl, accp);
                }
                const int n = (nt << 4) + (lane & 15);
                const float bias = gcn_b[h * H_DIM + n] + sage_l_b[h * H_DIM + n];
                #pragma unroll
                for (int v = 0; v < 8; ++v) {
                    const size_t o = hb + (size_t)(drow0 + v) * H_DIM + n;
                    xw16[o] = f2bf(accx[v]);
                    pre[o] = accp[v] + bias;
                }
            }
        }
        __syncthreads();  // protect LDS before restaging next head
    }
}

// Edge-wise GCN aggregation: pre[h][col] += xw[h][row] * dis[row]*dis[col]
__global__ void k_gcn_scatter(const int* __restrict__ uu_row, const int* __restrict__ uu_col,
                              const float* __restrict__ dis,
                              const unsigned short* __restrict__ xw16,
                              float* __restrict__ pre) {
    int t = blockIdx.x * blockDim.x + threadIdx.x;
    if (t >= E_TOT * 32) return;
    int e = t >> 5;
    int j = (t & 31) << 2;
    int r = uu_row[e];
    int c = uu_col[e];
    if (e + 64 < E_TOT)  // lookahead prefetch of a future random xw row
        __builtin_prefetch(xw16 + (size_t)uu_row[e + 64] * H_DIM + j, 0, 1);
    float w = dis[r] * dis[c];
    if (w == 0.f) return;  // zero-degree endpoints contribute nothing
    #pragma unroll
    for (int h = 0; h < NHEAD; ++h) {
        const unsigned short* sp = xw16 + ((size_t)h * U_TOT + r) * H_DIM + j;
        float* dp = pre + ((size_t)h * U_TOT + c) * H_DIM + j;
        unsigned long long pk = *(const unsigned long long*)sp;  // 4 bf16 in one 8B load
        atomicAdd(dp + 0, bf2f((unsigned short)(pk)) * w);
        atomicAdd(dp + 1, bf2f((unsigned short)(pk >> 16)) * w);
        atomicAdd(dp + 2, bf2f((unsigned short)(pk >> 32)) * w);
        atomicAdd(dp + 3, bf2f((unsigned short)(pk >> 48)) * w);
    }
}

// node = leaky(pre0) + leaky(pre1) stored into pre0; column sums for BN.
__global__ void k_combine(float* __restrict__ pre, float* __restrict__ bnsum,
                          float* __restrict__ bnsq) {
    const int j = threadIdx.x & 127;
    const int half = threadIdx.x >> 7;
    const int r0 = blockIdx.x * 100;
    float s = 0.f, s2 = 0.f;
    for (int i = r0 + half; i < r0 + 100; i += 2) {
        const size_t o = (size_t)i * H_DIM + j;
        float a = pre[o];
        float b = pre[(size_t)U_TOT * H_DIM + o];
        a = a > 0.f ? a : 0.3f * a;
        b = b > 0.f ? b : 0.3f * b;
        const float x = a + b;
        pre[o] = x;
        s += x;
        s2 += x * x;
    }
    __shared__ float sh[256];
    __shared__ float sh2[256];
    sh[threadIdx.x] = s;
    sh2[threadIdx.x] = s2;
    __syncthreads();
    if (half == 0) {
        atomicAdd(bnsum + j, sh[j] + sh[j + 128]);
        atomicAdd(bnsq + j, sh2[j] + sh2[j + 128]);
    }
}

__global__ void k_bn_out(const float* __restrict__ node, const float* __restrict__ bnsum,
                         const float* __restrict__ bnsq, const float* __restrict__ gamma,
                         const float* __restrict__ beta, float* __restrict__ out) {
    int t = blockIdx.x * blockDim.x + threadIdx.x;
    if (t >= U_TOT * H_DIM) return;
    int j = t & 127;
    const float inv = 1.0f / (float)U_TOT;
    float mean = bnsum[j] * inv;
    float var = fmaxf(bnsq[j] * inv - mean * mean, 0.f);
    out[t] = (node[t] - mean) * rsqrtf(var + 1e-5f) * gamma[j] + beta[j];
}

// --------------------------- launch ----------------------------------------

extern "C" void kernel_launch(void* const* d_in, const int* in_sizes, int n_in,
                              void* d_out, int out_size, void* d_ws, size_t ws_size,
                              hipStream_t stream) {
    const int*   no_Nidx   = (const int*)d_in[0];
    const float* u_feature = (const float*)d_in[1];
    const float* comment_x = (const float*)d_in[2];
    const int*   edge_uu   = (const int*)d_in[3];
    const int*   cu_src    = (const int*)d_in[4];
    const int*   cu_dst    = (const int*)d_in[5];
    const float* emb_table = (const float*)d_in[6];
    const float* e0        = (const float*)d_in[7];
    const float* e3        = (const float*)d_in[8];
    const float* e7        = (const float*)d_in[9];
    const float* e8        = (const float*)d_in[10];
    const float* e9        = (const float*)d_in[11];
    const float* gcn_w     = (const float*)d_in[12];
    const float* gcn_b     = (const float*)d_in[13];
    const float* sage_l_w  = (const float*)d_in[14];
    const float* sage_l_b  = (const float*)d_in[15];
    const float* sage_r_w  = (const float*)d_in[16];
    const float* bn_gamma  = (const float*)d_in[17];
    const float* bn_beta   = (const float*)d_in[18];
    float* out = (float*)d_out;

    char* ws = (char*)d_ws;
    float* deg = (float*)(ws + OFF_DEG);  // becomes dis after k_dis
    float* cnt = (float*)(ws + OFF_CNT);
    float* agg = (float*)(ws + OFF_AGG);
    float* bnsum = (float*)(ws + OFF_BNS);
    float* bnsq = (float*)(ws + OFF_BNS2);
    unsigned short* uf16 = (unsigned short*)(ws + OFF_UF16);
    unsigned short* ag16 = (unsigned short*)(ws + OFF_AG16);
    unsigned short* xw16 = (unsigned short*)(ws + OFF_XW16);
    float* pre = (float*)(ws + OFF_PRE);
    unsigned short* Bg = (unsigned short*)(ws + OFF_BG);
    unsigned short* Br = (unsigned short*)(ws + OFF_BR);
    unsigned short* Bl = (unsigned short*)(ws + OFF_BL);
    const int* uu_row = edge_uu;
    const int* uu_col = edge_uu + E_TOT;

    const int nz = (int)(ZERO_END / 4);
    k_zero<<<(nz + 255) / 256, 256, 0, stream>>>((float*)ws, nz);
    k_build_userF<<<(U_TOT + 255) / 256, 256, 0, stream>>>(
        u_feature, no_Nidx, emb_table, e0, e3, e7, e8, e9, uf16);
    {
        int tg = NHEAD * NT_TOT * KT_ED * 512;
        k_packB<<<(tg + 255) / 256, 256, 0, stream>>>(gcn_w, Bg, ED_DIM, KT_ED, tg);
        k_packB<<<(tg + 255) / 256, 256, 0, stream>>>(sage_r_w, Br, ED_DIM, KT_ED, tg);
        int tl = NHEAD * NT_TOT * KT_DC * 512;
        k_packB<<<(tl + 255) / 256, 256, 0, stream>>>(sage_l_w, Bl, DC_DIM, KT_DC, tl);
    }
    k_degcnt<<<(E_TOT + 255) / 256, 256, 0, stream>>>(uu_col, cu_dst, deg, cnt);
    k_dis<<<(U_TOT + 255) / 256, 256, 0, stream>>>(deg);
    k_agg_scatter<<<(E_TOT * 16 + 255) / 256, 256, 0, stream>>>(cu_src, cu_dst, comment_x, agg);
    k_agg_fin<<<(U_TOT * DC_DIM + 255) / 256, 256, 0, stream>>>(agg, cnt, ag16);
    k_gemm<<<(MT_TOT + 7) / 8, 256, 0, stream>>>(uf16, ag16, Bg, Br, Bl,
                                                 gcn_b, sage_l_b, xw16, pre);
    k_gcn_scatter<<<(E_TOT * 32 + 255) / 256, 256, 0, stream>>>(uu_row, uu_col, deg, xw16, pre);
    k_combine<<<U_TOT / 100, 256, 0, stream>>>(pre, bnsum, bnsq);
    k_bn_out<<<(U_TOT * H_DIM + 255) / 256, 256, 0, stream>>>(pre, bnsum, bnsq,
                                                              bn_gamma, bn_beta, out);
}
